// BIATT_7705171329546
// MI455X (gfx1250) — matrix operand order repaired and checked by reading, not compiled, not run on hardware
//
#include <hip/hip_runtime.h>

// ---------------------------------------------------------------------------
// BIATT on MI455X (gfx1250): algebraically collapsed bilinear attention.
// Heavy GEMMs: v_wmma_f32_16x16x32_bf16 (wave32, f32 accumulate).
// Data staging: Tensor Data Mover (tensor_load_to_lds + s_wait_tensorcnt),
// fragment loads as ds_load_b128 via contiguous per-lane LDS layout.
// ---------------------------------------------------------------------------

typedef __attribute__((ext_vector_type(16))) __bf16 v16bf;
typedef __attribute__((ext_vector_type(8)))  __bf16 v8bf;
typedef __attribute__((ext_vector_type(8)))  float  v8f;
typedef __attribute__((ext_vector_type(4)))  unsigned int u32x4;
typedef __attribute__((ext_vector_type(8)))  int i32x8;
typedef __attribute__((ext_vector_type(4)))  int i32x4;

constexpr int NB  = 16;
constexpr int NLC = 1024;
constexpr int NLP = 1024;
constexpr int ND  = 128;
constexpr int NH  = 4;
constexpr int NPF = 1024;

__device__ static inline unsigned ldsaddr(const void* p) {
  return (unsigned)(unsigned long long)(uintptr_t)p;   // LDS aperture: low 32 bits
}

// ---- Tensor Data Mover: 2D tile load, D# built per CDNA5 ISA ch.8 ----------
// w0 carries data_size / pad fields of D# group1 word0.
__device__ static inline void tdm_load_2d(unsigned lds_addr, const void* gptr,
                                          unsigned w0, unsigned td0, unsigned td1,
                                          unsigned tile0, unsigned tile1,
                                          unsigned stride0) {
  const unsigned long long ga = (unsigned long long)(uintptr_t)gptr;
  u32x4 g0 = { 1u,                                   // count=1, user descriptor
               lds_addr,
               (unsigned)ga,
               (unsigned)((ga >> 32) & 0x01FFFFFFu) | (2u << 30) };  // type=2
  i32x8 g1 = { (int)w0,                              // data_size | pad ctl
               (int)(td0 << 16),                     // tensor_dim0[15:0]
               (int)((td0 >> 16) | (td1 << 16)),     // td0 hi | tensor_dim1 lo
               (int)((td1 >> 16) | (tile0 << 16)),   // td1 hi | tile_dim0
               (int)(tile1 & 0xFFFFu),               // tile_dim1 (tile_dim2=0)
               (int)stride0,                         // tensor_dim0_stride lo
               0, 0 };
  i32x4 gz4 = { 0, 0, 0, 0 };
  i32x8 gz8 = { 0, 0, 0, 0, 0, 0, 0, 0 };
  __builtin_amdgcn_tensor_load_to_lds(g0, g1, gz4, gz4, gz8, 0);
}

// ---- WMMA fragment loaders: two ds_load_b128 per fragment ------------------
// A (16x32 bf16, row-major LDS, ld multiple of 8): lane holds M=lane&15,
// K = kb..kb+7 and kb+16..kb+23 with kb = 8*(lane>>4)  -> contiguous runs.
__device__ static inline v16bf ldA(const __bf16* base, int ld, int lane) {
  const int m  = lane & 15;
  const int kb = (lane >> 4) << 3;
  const v8bf lo = *(const v8bf*)(base + m * ld + kb);
  const v8bf hi = *(const v8bf*)(base + m * ld + kb + 16);
  return __builtin_shufflevector(lo, hi, 0, 1, 2, 3, 4, 5, 6, 7,
                                 8, 9, 10, 11, 12, 13, 14, 15);
}
// B (KxN) stored TRANSPOSED in LDS as [n][k]: lane holds N=(lane&15)+n0,
// K = kb..kb+7 and kb+16..kb+23 with kb = 8*(lane>>4).
__device__ static inline v16bf ldBT(const __bf16* base, int ld, int n0, int lane) {
  const int n  = (lane & 15) + n0;
  const int kb = (lane >> 4) << 3;
  const v8bf lo = *(const v8bf*)(base + n * ld + kb);
  const v8bf hi = *(const v8bf*)(base + n * ld + kb + 16);
  return __builtin_shufflevector(lo, hi, 0, 1, 2, 3, 4, 5, 6, 7,
                                 8, 9, 10, 11, 12, 13, 14, 15);
}

__device__ static inline float fast_tanh(float x) {
  return 1.f - 2.f / (__expf(2.f * x) + 1.f);
}

// ---------------------------------------------------------------------------
// K0: collapse c_param through U, Wh_c, W_c2p, W_comb_c -> four [H,128] vecs
// ---------------------------------------------------------------------------
__global__ __launch_bounds__(256) void k_prep(
    const float* __restrict__ c_param, const float* __restrict__ U,
    const float* __restrict__ Wh_c, const float* __restrict__ W_c2p,
    const float* __restrict__ W_comb_c,
    float* __restrict__ cu, float* __restrict__ whc,
    float* __restrict__ wc2p, float* __restrict__ wcc) {
  __shared__ float cp[ND];
  if (threadIdx.x < ND) cp[threadIdx.x] = c_param[threadIdx.x];
  __syncthreads();
  for (int i = threadIdx.x; i < 4 * NH * ND; i += 256) {
    const int arr = i >> 9;
    const int h   = (i >> 7) & 3;
    const int e   = i & 127;
    const float* W = (arr == 0) ? U : (arr == 1) ? Wh_c : (arr == 2) ? W_c2p : W_comb_c;
    float a0 = 0.f;
    for (int d = 0; d < ND; ++d) a0 += cp[d] * W[(h * ND + d) * ND + e];
    float* dst = (arr == 0) ? cu : (arr == 1) ? whc : (arr == 2) ? wc2p : wcc;
    dst[h * ND + e] = a0;
  }
}

// ---------------------------------------------------------------------------
// K1: pv = p @ p_param (bf16 WMMA, K=1024). Block tile 64x128, 8 waves,
// 4 accumulators/wave. p tiles staged by TDM (double-buffered).
// ---------------------------------------------------------------------------
__global__ __launch_bounds__(256) void k_pv_gemm(
    const float* __restrict__ p, const float* __restrict__ p_param,
    __bf16* __restrict__ pv) {
  __shared__ __align__(16) float  Pst[2][64][32];   // TDM staging (f32)
  __shared__ __align__(16) __bf16 At[64][40];
  __shared__ __align__(16) __bf16 BtT[128][40];     // p_param chunk, [n][k]
  const int t = threadIdx.x, lane = t & 31, wave = t >> 5;
  const int b  = blockIdx.x >> 4;            // / (LP/64)
  const int m0 = (blockIdx.x & 15) << 6;
  const float* pb = p + ((size_t)b * NLP + m0) * NPF;

  if (t < 32)   // wave 0 issues chunk 0
    tdm_load_2d(ldsaddr(&Pst[0][0][0]), pb, (2u << 16), NPF, 64, 32, 64, NPF);

  v8f acc[4] = {};
  for (int c = 0; c < 32; ++c) {
    if (t < 32) {
      if (c + 1 < 32) {
        tdm_load_2d(ldsaddr(&Pst[(c + 1) & 1][0][0]), pb + (c + 1) * 32,
                    (2u << 16), NPF, 64, 32, 64, NPF);
        __builtin_amdgcn_s_wait_tensorcnt(1);   // allow the just-issued one
      } else {
        __builtin_amdgcn_s_wait_tensorcnt(0);
      }
    }
    __syncthreads();
    for (int i = t; i < 64 * 32; i += 256) {           // f32 -> bf16
      const int r = i >> 5, cc = i & 31;
      At[r][cc] = (__bf16)Pst[c & 1][r][cc];
    }
    for (int i = t; i < 32 * 128; i += 256) {          // stage B transposed
      const int k = i >> 7, n = i & 127;
      BtT[n][k] = (__bf16)p_param[(c * 32 + k) * ND + n];
    }
    __syncthreads();
    const v16bf bb = ldBT(&BtT[0][0], 40, wave * 16, lane);
#pragma unroll
    for (int s4 = 0; s4 < 4; ++s4) {
      const v16bf a = ldA(&At[s4 * 16][0], 40, lane);
      acc[s4] = __builtin_amdgcn_wmma_f32_16x16x32_bf16(false, a, false, bb,
                                                        (short)0, acc[s4],
                                                        false, false);
    }
    __syncthreads();
  }
  const int n = (lane & 15) + wave * 16;
#pragma unroll
  for (int s4 = 0; s4 < 4; ++s4)
#pragma unroll
    for (int r = 0; r < 8; ++r) {
      const int m = s4 * 16 + r + ((lane >> 4) << 3);
      pv[((size_t)b * NLP + m0 + m) * ND + n] = (__bf16)acc[s4][r];
    }
}

// ---------------------------------------------------------------------------
// K2: per (b,h,128-row chunk): s = pv.cu[h]; q = tanh(pv@W_p2c+b).wa_c2;
// hp1 = tanh(pv@Wh_p+b).wa_p1. W staged once/pass; pv tiles via TDM with
// pad_enable so rows land at the padded 136-element stride directly.
// ---------------------------------------------------------------------------
__global__ __launch_bounds__(256) void k_transform(
    const __bf16* __restrict__ pv,
    const float* __restrict__ W_p2c, const float* __restrict__ b_p2c,
    const float* __restrict__ Wh_p,  const float* __restrict__ bh_p,
    const float* __restrict__ Wa_c,  const float* __restrict__ Wa_p,
    const float* __restrict__ cu,
    float* __restrict__ s, float* __restrict__ q, float* __restrict__ hp1) {
  __shared__ __align__(16) __bf16 Wt[128][136];      // [n][k], padded
  __shared__ __align__(16) __bf16 pvT[16][136];      // TDM pad: 64DW + 4DW
  __shared__ float part[16][132];
  __shared__ float ps[16][16];
  const int t = threadIdx.x, lane = t & 31, wave = t >> 5;
  const int mc = blockIdx.x & 7;             // 8 chunks of 128 rows
  const int bh = blockIdx.x >> 3;
  const int h  = bh & 3, b = bh >> 2;
  const int mbase = mc << 7;
  // D# w0: data_size=2B, pad_enable, interval=64 DW (code 5), amount=4 DW (code 3)
  const unsigned pvW0 = (1u << 16) | (1u << 20) | (5u << 22) | (3u << 25);

  for (int pass = 0; pass < 2; ++pass) {
    const float* W    = pass ? (Wh_p + h * ND * ND) : (W_p2c + h * ND * ND);
    const float* bias = pass ? (bh_p + h * ND)      : (b_p2c + h * ND);
    const float* wa   = pass ? (Wa_p + h * 2 * ND)  : (Wa_c + h * 2 * ND + ND);
    float* outv       = pass ? hp1 : q;
    __syncthreads();
    for (int i = t; i < ND * ND; i += 256) {           // stage W transposed
      const int k = i >> 7, n = i & 127;
      Wt[n][k] = (__bf16)W[k * ND + n];
    }
    __syncthreads();
    for (int mt = 0; mt < 8; ++mt) {
      const int m0 = mbase + mt * 16;
      if (t < 32) {
        tdm_load_2d(ldsaddr(&pvT[0][0]), pv + ((size_t)b * NLP + m0) * ND,
                    pvW0, ND, 16, ND, 16, ND);
        __builtin_amdgcn_s_wait_tensorcnt(0);
      }
      __syncthreads();
      if (pass == 0) {               // s[b,h,m] = pv[m,:].cu[h] (deterministic)
        const int m = t >> 4, seg = t & 15;
        float a0 = 0.f;
        for (int e = seg * 8; e < seg * 8 + 8; ++e)
          a0 += (float)pvT[m][e] * cu[h * ND + e];
        ps[m][seg] = a0;
        __syncthreads();
        if (t < 16) {
          float a0 = 0.f;
          for (int g = 0; g < 16; ++g) a0 += ps[t][g];
          s[(size_t)bh * NLP + m0 + t] = a0;
        }
      }
      v8f acc = {};
#pragma unroll
      for (int k0 = 0; k0 < ND; k0 += 32) {
        const v16bf a  = ldA(&pvT[0][k0], 136, lane);
        const v16bf bb = ldBT(&Wt[0][k0], 136, wave * 16, lane);
        acc = __builtin_amdgcn_wmma_f32_16x16x32_bf16(false, a, false, bb,
                                                      (short)0, acc,
                                                      false, false);
      }
      const int n = (lane & 15) + wave * 16;
      const float bn = bias[n], wn = wa[n];
#pragma unroll
      for (int r = 0; r < 8; ++r)
        part[r + ((lane >> 4) << 3)][n] = tanhf(acc[r] + bn) * wn;
      __syncthreads();
      if (t < 16) {
        float a0 = 0.f;
        for (int n2 = 0; n2 < ND; ++n2) a0 += part[t][n2];
        outv[(size_t)bh * NLP + m0 + t] = a0;
      }
      __syncthreads();               // before next TDM overwrites pvT
    }
  }
}

// ---------------------------------------------------------------------------
// K3a: c-side scalars: hc1 = tanh(c_l*whc+bh_c).wa_c1 ; r = tanh(c_l*wc2p+b).wa_p2
// ---------------------------------------------------------------------------
__global__ __launch_bounds__(256) void k_cside(
    const float* __restrict__ c, const float* __restrict__ whc,
    const float* __restrict__ bh_c, const float* __restrict__ wc2p,
    const float* __restrict__ b_c2p, const float* __restrict__ Wa_c,
    const float* __restrict__ Wa_p, float* __restrict__ hc1,
    float* __restrict__ rr) {
  const int idx = blockIdx.x * 256 + threadIdx.x;   // ((b*H+h)*LC + l)
  const int l = idx & (NLC - 1);
  const int h = (idx >> 10) & (NH - 1);
  const int b = idx >> 12;
  const float cl = c[b * NLC + l];
  float a1 = 0.f, a2 = 0.f;
  for (int e = 0; e < ND; ++e) {
    a1 += tanhf(cl * whc[h * ND + e] + bh_c[h * ND + e]) * Wa_c[h * 2 * ND + e];
    a2 += tanhf(cl * wc2p[h * ND + e] + b_c2p[h * ND + e]) * Wa_p[h * 2 * ND + ND + e];
  }
  hc1[idx] = a1;
  rr[idx] = a2;
}

// ---------------------------------------------------------------------------
// K3b: score_c[l] = hc1[l] + cmask_l * sum_m tanh(c_l*s_m)*pmask_m*q_m + ba_c
// ---------------------------------------------------------------------------
__global__ __launch_bounds__(256) void k_score_c(
    const float* __restrict__ c, const float* __restrict__ c_mask,
    const float* __restrict__ p_mask, const float* __restrict__ s,
    const float* __restrict__ q, const float* __restrict__ hc1,
    const float* __restrict__ ba_c, float* __restrict__ scc) {
  __shared__ float ss[NLP], qq[NLP];
  const int t = threadIdx.x;
  const int chunk = blockIdx.x & 3;
  const int bh = blockIdx.x >> 2, b = bh >> 2, h = bh & 3;
  for (int i = t; i < NLP; i += 256) {
    ss[i] = s[(size_t)bh * NLP + i];
    qq[i] = q[(size_t)bh * NLP + i] * p_mask[b * NLP + i];
  }
  __syncthreads();
  const int l = chunk * 256 + t;
  const float cl = c[b * NLC + l];
  float acc = 0.f;
#pragma unroll 4
  for (int m = 0; m < NLP; ++m) acc += fast_tanh(cl * ss[m]) * qq[m];
  const int idx = bh * NLC + l;
  scc[idx] = hc1[idx] + c_mask[b * NLC + l] * acc + ba_c[h];
}

// ---------------------------------------------------------------------------
// K3c: score_p[m] = hp1[m] + pmask_m * sum_l tanh(c_l*s_m)*cmask_l*r_l + ba_p
// ---------------------------------------------------------------------------
__global__ __launch_bounds__(256) void k_score_p(
    const float* __restrict__ c, const float* __restrict__ c_mask,
    const float* __restrict__ p_mask, const float* __restrict__ s,
    const float* __restrict__ rr, const float* __restrict__ hp1,
    const float* __restrict__ ba_p, float* __restrict__ scp) {
  __shared__ float cl_[NLC], rm[NLC];
  const int t = threadIdx.x;
  const int chunk = blockIdx.x & 3;
  const int bh = blockIdx.x >> 2, b = bh >> 2, h = bh & 3;
  for (int i = t; i < NLC; i += 256) {
    cl_[i] = c[b * NLC + i];
    rm[i]  = rr[(size_t)bh * NLC + i] * c_mask[b * NLC + i];
  }
  __syncthreads();
  const int m = chunk * 256 + t;
  const float sm = s[(size_t)bh * NLP + m];
  float acc = 0.f;
#pragma unroll 4
  for (int l = 0; l < NLC; ++l) acc += fast_tanh(cl_[l] * sm) * rm[l];
  const int idx = bh * NLP + m;
  scp[idx] = hp1[idx] + p_mask[b * NLP + m] * acc + ba_p[h];
}

// ---------------------------------------------------------------------------
// K4: masked softmaxes; alpha[b,h] = sum_l c_l*att_c ; pf[b,h,:] = att_p @ pv
// ---------------------------------------------------------------------------
__global__ __launch_bounds__(256) void k_softmax(
    const float* __restrict__ c, const float* __restrict__ c_mask,
    const float* __restrict__ p_mask, const __bf16* __restrict__ pv,
    const float* __restrict__ scc, const float* __restrict__ scp,
    float* __restrict__ alpha, float* __restrict__ pf) {
  __shared__ float buf[NLP];
  __shared__ float red[256];
  __shared__ float pfh[2][ND];
  const int t = threadIdx.x;
  const int bh = blockIdx.x, b = bh >> 2;

  // ---- c side: softmax + alpha ----
  for (int i = t; i < NLC; i += 256) buf[i] = scc[(size_t)bh * NLC + i];
  __syncthreads();
  float lm = -3.4e38f;
  for (int i = t; i < NLC; i += 256) lm = fmaxf(lm, buf[i]);
  red[t] = lm; __syncthreads();
  for (int st = 128; st > 0; st >>= 1) { if (t < st) red[t] = fmaxf(red[t], red[t + st]); __syncthreads(); }
  const float mxc = red[0]; __syncthreads();
  float ls = 0.f;
  for (int i = t; i < NLC; i += 256) {
    const float e = __expf(buf[i] - mxc) * c_mask[b * NLC + i];
    buf[i] = e; ls += e;
  }
  red[t] = ls; __syncthreads();
  for (int st = 128; st > 0; st >>= 1) { if (t < st) red[t] += red[t + st]; __syncthreads(); }
  const float invc = 1.f / (red[0] + 1e-6f); __syncthreads();
  float la = 0.f;
  for (int i = t; i < NLC; i += 256) la += c[b * NLC + i] * buf[i];
  red[t] = la; __syncthreads();
  for (int st = 128; st > 0; st >>= 1) { if (t < st) red[t] += red[t + st]; __syncthreads(); }
  if (t == 0) alpha[bh] = red[0] * invc;
  __syncthreads();

  // ---- p side: softmax + pf GEMV ----
  for (int i = t; i < NLP; i += 256) buf[i] = scp[(size_t)bh * NLP + i];
  __syncthreads();
  lm = -3.4e38f;
  for (int i = t; i < NLP; i += 256) lm = fmaxf(lm, buf[i]);
  red[t] = lm; __syncthreads();
  for (int st = 128; st > 0; st >>= 1) { if (t < st) red[t] = fmaxf(red[t], red[t + st]); __syncthreads(); }
  const float mxp = red[0]; __syncthreads();
  ls = 0.f;
  for (int i = t; i < NLP; i += 256) {
    const float e = __expf(buf[i] - mxp) * p_mask[b * NLP + i];
    buf[i] = e; ls += e;
  }
  red[t] = ls; __syncthreads();
  for (int st = 128; st > 0; st >>= 1) { if (t < st) red[t] += red[t + st]; __syncthreads(); }
  const float invp = 1.f / (red[0] + 1e-6f); __syncthreads();
  for (int i = t; i < NLP; i += 256) buf[i] *= invp;
  __syncthreads();
  const int e = t & 127, half = t >> 7;
  float acc = 0.f;
  for (int m = half * 512; m < half * 512 + 512; ++m)
    acc += buf[m] * (float)pv[((size_t)b * NLP + m) * ND + e];
  pfh[half][e] = acc; __syncthreads();
  if (t < ND) pf[(size_t)bh * ND + t] = pfh[0][t] + pfh[1][t];
}

// ---------------------------------------------------------------------------
// K5: cf_final = sum_h alpha_h*wcc[h] + b ; pf_final = pf@W_comb_p + b ;
//     out[b,i,j] = cf_final[i]*pf_final[j]
// ---------------------------------------------------------------------------
__global__ __launch_bounds__(256) void k_final(
    const float* __restrict__ alpha, const float* __restrict__ wcc,
    const float* __restrict__ b_comb_c, const float* __restrict__ pf,
    const float* __restrict__ W_comb_p, const float* __restrict__ b_comb_p,
    float* __restrict__ out) {
  __shared__ float cf[ND], pff[ND];
  const int b = blockIdx.x, t = threadIdx.x;
  if (t < ND) {
    float a0 = b_comb_c[t];
#pragma unroll
    for (int h = 0; h < NH; ++h) a0 += alpha[b * NH + h] * wcc[h * ND + t];
    cf[t] = a0;
  } else {
    const int j = t - ND;
    float a1 = b_comb_p[j];
    for (int i = 0; i < NH * ND; ++i)
      a1 += pf[(size_t)b * NH * ND + i] * W_comb_p[i * ND + j];
    pff[j] = a1;
  }
  __syncthreads();
  for (int idx = t; idx < ND * ND; idx += 256)
    out[(size_t)b * ND * ND + idx] = cf[idx >> 7] * pff[idx & 127];
}

// ---------------------------------------------------------------------------
extern "C" void kernel_launch(void* const* d_in, const int* in_sizes, int n_in,
                              void* d_out, int out_size, void* d_ws, size_t ws_size,
                              hipStream_t stream) {
  const float* c        = (const float*)d_in[0];
  const float* c_mask   = (const float*)d_in[1];
  const float* p        = (const float*)d_in[2];
  const float* p_mask   = (const float*)d_in[3];
  const float* c_param  = (const float*)d_in[4];
  const float* p_param  = (const float*)d_in[5];
  const float* U        = (const float*)d_in[6];
  const float* W_p2c    = (const float*)d_in[7];
  const float* b_p2c    = (const float*)d_in[8];
  const float* W_c2p    = (const float*)d_in[9];
  const float* b_c2p    = (const float*)d_in[10];
  const float* Wh_c     = (const float*)d_in[11];
  const float* bh_c     = (const float*)d_in[12];
  const float* Wh_p     = (const float*)d_in[13];
  const float* bh_p     = (const float*)d_in[14];
  const float* Wa_c     = (const float*)d_in[15];
  const float* ba_c     = (const float*)d_in[16];
  const float* Wa_p     = (const float*)d_in[17];
  const float* ba_p     = (const float*)d_in[18];
  const float* W_comb_c = (const float*)d_in[19];
  const float* b_comb_c = (const float*)d_in[20];
  const float* W_comb_p = (const float*)d_in[21];
  const float* b_comb_p = (const float*)d_in[22];
  float* out = (float*)d_out;

  // ---- workspace layout ----
  __bf16* pv = (__bf16*)d_ws;                                   // B*LP*D bf16
  float* fws = (float*)((char*)d_ws + (size_t)NB * NLP * ND * sizeof(__bf16));
  float* cu    = fws;                 // H*D
  float* whc   = cu    + NH * ND;
  float* wc2p  = whc   + NH * ND;
  float* wcc   = wc2p  + NH * ND;
  float* s     = wcc   + NH * ND;     // B*H*LP
  float* q     = s     + NB * NH * NLP;
  float* hp1   = q     + NB * NH * NLP;
  float* hc1   = hp1   + NB * NH * NLP;   // B*H*LC
  float* rr    = hc1   + NB * NH * NLC;
  float* scc   = rr    + NB * NH * NLC;
  float* scp   = scc   + NB * NH * NLC;
  float* alpha = scp   + NB * NH * NLP;   // B*H
  float* pf    = alpha + NB * NH;         // B*H*D

  k_prep<<<1, 256, 0, stream>>>(c_param, U, Wh_c, W_c2p, W_comb_c,
                                cu, whc, wc2p, wcc);
  k_pv_gemm<<<NB * (NLP / 64), 256, 0, stream>>>(p, p_param, pv);
  k_transform<<<NB * NH * 8, 256, 0, stream>>>(
      pv, W_p2c, b_p2c, Wh_p, bh_p, Wa_c, Wa_p, cu, s, q, hp1);
  k_cside<<<(NB * NH * NLC) / 256, 256, 0, stream>>>(
      c, whc, bh_c, wc2p, b_c2p, Wa_c, Wa_p, hc1, rr);
  k_score_c<<<NB * NH * (NLC / 256), 256, 0, stream>>>(
      c, c_mask, p_mask, s, q, hc1, ba_c, scc);
  k_score_p<<<NB * NH * (NLP / 256), 256, 0, stream>>>(
      c, c_mask, p_mask, s, rr, hp1, ba_p, scp);
  k_softmax<<<NB * NH, 256, 0, stream>>>(
      c, c_mask, p_mask, pv, scc, scp, alpha, pf);
  k_final<<<NB, 256, 0, stream>>>(
      alpha, wcc, b_comb_c, pf, W_comb_p, b_comb_p, out);
}
